// Pure_MZI_Net_14877766713928
// MI455X (gfx1250) — matrix operand order repaired and checked by reading, not compiled, not run on hardware
//
#include <hip/hip_runtime.h>

// MZI photonic net: 3x (dense 256x256 complex unitary GEMM + fused EO nonlinearity),
// then intensity + L2 row-normalize.  GEMMs run on V_WMMA_F32_16X16X32_F16.

typedef __attribute__((ext_vector_type(16))) _Float16 v16h;
typedef __attribute__((ext_vector_type(8)))  float    v8f;

#define N_WG   256
#define BATCH  8192

// ---------------------------------------------------------------------------
// Kernel 1: compose the 256 MZI layers (+gamma) of each mesh into a 256x256
// complex unitary, held as f32 column-slices in LDS, then emit it quantized to
// f16 directly in the WMMA B-operand register layout:
//   Bpack[mesh][comp(r/i)][kb(=K/32)][nt(=N/16)][lane][16 halves]
//   lane = (K half-block)*16 + n%16 ; half e of lane -> K = kb*32 + (lane/16)*16 + e
// grid: 48 blocks (3 meshes x 16 column-blocks of 16), 256 threads.
// ---------------------------------------------------------------------------
__global__ __launch_bounds__(256)
void compose_mesh(const float* __restrict__ theta,
                  const float* __restrict__ phi,
                  const float* __restrict__ gamma,
                  _Float16*    __restrict__ Bpack)
{
    __shared__ float Ur[256][16];   // 16 KB
    __shared__ float Ui[256][16];   // 16 KB
    __shared__ float Cf[128][8];    //  4 KB : per-pair 2x2 complex coefficients

    const int t    = threadIdx.x;
    const int mesh = blockIdx.x >> 4;   // 0..2
    const int cb   = blockIdx.x & 15;   // which 16-column slice of U
    const int c0   = cb * 16;

    // identity init of the owned column slice
    #pragma unroll
    for (int it = 0; it < 16; ++it) {
        int idx = it * 256 + t;
        int r = idx >> 4, c = idx & 15;
        Ur[r][c] = (r == (c0 + c)) ? 1.0f : 0.0f;
        Ui[r][c] = 0.0f;
    }
    __syncthreads();

    for (int l = 0; l < 256; ++l) {
        const int sh = l & 1;
        // phase 1: 2x2 coefficients (threads 0..127, one MZI each)
        if (t < 128) {
            const int k = t;
            const bool active = (sh == 0) || (k < 127);   // wrap pair masked on odd layers
            float a_r, a_i, b_r, b_i, c_r, c_i, d_r, d_i;
            if (active) {
                const float th = theta[(mesh * 256 + l) * 128 + k];
                const float pp = phi  [(mesh * 256 + l) * 128 + k];
                float s, c;  __sincosf(0.5f * th, &s, &c);
                float epi, epr; __sincosf(pp, &epi, &epr);        // ep = e^{i phi}
                const float prer = -s, prei = c;                  // pre = i e^{i th/2}
                const float per = prer * epr - prei * epi;        // pre*ep
                const float pei = prer * epi + prei * epr;
                a_r = per  * s;  a_i = pei  * s;                  // y0 = a x0 + b x1
                b_r = prer * c;  b_i = prei * c;
                c_r = per  * c;  c_i = pei  * c;                  // y1 = c x0 + d x1
                d_r = -prer * s; d_i = -prei * s;
            } else {
                a_r = 1.f; a_i = 0.f; b_r = 0.f; b_i = 0.f;
                c_r = 0.f; c_i = 0.f; d_r = 1.f; d_i = 0.f;
            }
            Cf[k][0]=a_r; Cf[k][1]=a_i; Cf[k][2]=b_r; Cf[k][3]=b_i;
            Cf[k][4]=c_r; Cf[k][5]=c_i; Cf[k][6]=d_r; Cf[k][7]=d_i;
        }
        __syncthreads();
        // phase 2: apply row-pair rotation to the column slice (exclusive row ownership)
        #pragma unroll
        for (int it = 0; it < 8; ++it) {
            int idx = it * 256 + t;
            int k = idx >> 4, col = idx & 15;
            int r0 = (2 * k + sh) & 255;
            int r1 = (2 * k + 1 + sh) & 255;
            float x0r = Ur[r0][col], x0i = Ui[r0][col];
            float x1r = Ur[r1][col], x1i = Ui[r1][col];
            float a_r=Cf[k][0], a_i=Cf[k][1], b_r=Cf[k][2], b_i=Cf[k][3];
            float c_r=Cf[k][4], c_i=Cf[k][5], d_r=Cf[k][6], d_i=Cf[k][7];
            Ur[r0][col] = a_r*x0r - a_i*x0i + b_r*x1r - b_i*x1i;
            Ui[r0][col] = a_r*x0i + a_i*x0r + b_r*x1i + b_i*x1r;
            Ur[r1][col] = c_r*x0r - c_i*x0i + d_r*x1r - d_i*x1i;
            Ui[r1][col] = c_r*x0i + c_i*x0r + d_r*x1i + d_i*x1r;
        }
        __syncthreads();
    }

    // output phases: row r scaled by e^{i gamma[r]}
    #pragma unroll
    for (int it = 0; it < 16; ++it) {
        int idx = it * 256 + t;
        int r = idx >> 4, c = idx & 15;
        float sg, cg; __sincosf(gamma[mesh * 256 + r], &sg, &cg);
        float ur = Ur[r][c], ui = Ui[r][c];
        Ur[r][c] = ur * cg - ui * sg;
        Ui[r][c] = ur * sg + ui * cg;
    }
    __syncthreads();

    // emit f16 in WMMA-B register layout; this slice is 16 K-values = half of
    // K-block kb = cb/2, selecting lane half-group hb = cb&1.
    const int kb = cb >> 1;
    const int hb = cb & 1;
    _Float16* baseR = Bpack + (mesh * 2 + 0) * 65536;
    _Float16* baseI = Bpack + (mesh * 2 + 1) * 65536;
    #pragma unroll
    for (int chunk = 0; chunk < 16; ++chunk) {
        int off    = chunk * 256 + t;        // 0..4095
        int idx16  = off & 15;               // K half element
        int lane15 = (off >> 4) & 15;        // n % 16
        int nt     = off >> 8;               // N tile
        int lane   = hb * 16 + lane15;
        int n      = nt * 16 + lane15;
        int addr   = ((kb * 16 + nt) * 32 + lane) * 16 + idx16;
        baseR[addr] = (_Float16)Ur[n][idx16];
        baseI[addr] = (_Float16)Ui[n][idx16];
    }
}

// ---------------------------------------------------------------------------
// A-operand fetch: row-major f32 -> v16h in the 16-bit WMMA A layout.
// lane holds row m = lane%16; halves 0..7 = K [k1,k1+8), halves 8..15 = K [k2,k2+8)
// ---------------------------------------------------------------------------
__device__ __forceinline__ v16h cvt_a(const float* __restrict__ rowp, int k1, int k2)
{
    float4 u0 = *(const float4*)(rowp + k1);
    float4 u1 = *(const float4*)(rowp + k1 + 4);
    float4 u2 = *(const float4*)(rowp + k2);
    float4 u3 = *(const float4*)(rowp + k2 + 4);
    v16h a;
    a[0]=(_Float16)u0.x;  a[1]=(_Float16)u0.y;  a[2]=(_Float16)u0.z;  a[3]=(_Float16)u0.w;
    a[4]=(_Float16)u1.x;  a[5]=(_Float16)u1.y;  a[6]=(_Float16)u1.z;  a[7]=(_Float16)u1.w;
    a[8]=(_Float16)u2.x;  a[9]=(_Float16)u2.y;  a[10]=(_Float16)u2.z; a[11]=(_Float16)u2.w;
    a[12]=(_Float16)u3.x; a[13]=(_Float16)u3.y; a[14]=(_Float16)u3.z; a[15]=(_Float16)u3.w;
    return a;
}

#define WMMA_F16(A, B, C) \
    __builtin_amdgcn_wmma_f32_16x16x32_f16(false, (A), false, (B), (short)0, (C), false, false)

// ---------------------------------------------------------------------------
// Kernel 2: Y[b,n] = sum_c U[n][c] * X[b,c]  (complex, f16 WMMA, f32 accum)
// fused epilogue: EO nonlinearity (final_pass==0) or intensity (final_pass==1).
// grid: 512 blocks (batch tiles of 16) x 256 threads; wave w does n-tiles {w, w+8}.
// ---------------------------------------------------------------------------
__global__ __launch_bounds__(256)
void mesh_gemm(const float* __restrict__ Xr, const float* __restrict__ Xi,
               const _Float16* __restrict__ Bmesh,
               float* __restrict__ Or, float* __restrict__ Oi, int final_pass)
{
    const int t    = threadIdx.x;
    const int lane = t & 31;
    const int w    = t >> 5;
    const int bt   = blockIdx.x;
    const int nt0  = w, nt1 = w + 8;
    const int m    = lane & 15;            // A row within tile / C column
    const int g    = lane >> 4;            // K half-group
    const int row  = bt * 16 + m;
    const float* xrp = Xr + row * 256;
    const float* xip = Xi + row * 256;
    const _Float16* BR = Bmesh;
    const _Float16* BI = Bmesh + 65536;

    v8f cr0 = {}, ci0 = {}, cr1 = {}, ci1 = {};

    #pragma unroll
    for (int kb = 0; kb < 8; ++kb) {
        const int k1 = kb * 32 + g * 16;
        const int k2 = k1 + 8;
        v16h ar  = cvt_a(xrp, k1, k2);
        v16h ai  = cvt_a(xip, k1, k2);
        v16h nai = -ai;                                   // f16 WMMA has no A-neg modifier
        const int bo0 = ((kb * 16 + nt0) * 32 + lane) * 16;
        const int bo1 = ((kb * 16 + nt1) * 32 + lane) * 16;
        v16h br0 = *(const v16h*)(BR + bo0);
        v16h bi0 = *(const v16h*)(BI + bo0);
        v16h br1 = *(const v16h*)(BR + bo1);
        v16h bi1 = *(const v16h*)(BI + bo1);
        // yr += Ar*Ur - Ai*Ui ; yi += Ar*Ui + Ai*Ur
        cr0 = WMMA_F16(ar,  br0, cr0);
        cr0 = WMMA_F16(nai, bi0, cr0);
        ci0 = WMMA_F16(ar,  bi0, ci0);
        ci0 = WMMA_F16(ai,  br0, ci0);
        cr1 = WMMA_F16(ar,  br1, cr1);
        cr1 = WMMA_F16(nai, bi1, cr1);
        ci1 = WMMA_F16(ar,  bi1, ci1);
        ci1 = WMMA_F16(ai,  br1, ci1);
    }

    // C layout: lane holds column n = lane%16; VGPR j -> row (lane/16)*8 + j
    #pragma unroll
    for (int j = 0; j < 8; ++j) {
        const int brow = bt * 16 + g * 8 + j;
        #pragma unroll
        for (int tile = 0; tile < 2; ++tile) {
            const int   col = (tile ? nt1 : nt0) * 16 + m;
            const float yr  = tile ? cr1[j] : cr0[j];
            const float yi  = tile ? ci1[j] : ci0[j];
            if (final_pass) {
                Or[brow * 256 + col] = yr * yr + yi * yi;            // intensity
            } else {
                // phase = 0.5*G*|y|^2 + 0.5*phi_b ; out = sqrt(1-a)*cos(p)*e^{-ip}*y
                const float p = 0.07853981634f * (yr * yr + yi * yi) + 1.57079632679f;
                float sp, cp; __sincosf(p, &sp, &cp);
                const float f = 0.9486832981f * cp;
                Or[brow * 256 + col] = f * (cp * yr + sp * yi);
                Oi[brow * 256 + col] = f * (cp * yi - sp * yr);
            }
        }
    }
}

// ---------------------------------------------------------------------------
// Kernel 3: per-row L2 normalize of intensities, emit first 10 columns.
// One wave32 per batch row; 1024 blocks x 256 threads.
// ---------------------------------------------------------------------------
__global__ __launch_bounds__(256)
void normalize_k(const float* __restrict__ I, float* __restrict__ out)
{
    const int gid  = blockIdx.x * 256 + threadIdx.x;
    const int row  = gid >> 5;
    const int lane = gid & 31;
    const float* r = I + row * 256;
    float s = 0.f;
    #pragma unroll
    for (int j = 0; j < 8; ++j) { float v = r[lane + 32 * j]; s += v * v; }
    #pragma unroll
    for (int off = 16; off; off >>= 1) s += __shfl_xor(s, off, 32);
    const float inv = 1.0f / fmaxf(sqrtf(s), 1e-12f);
    if (lane < 10) out[row * 10 + lane] = r[lane] * inv;
}

// ---------------------------------------------------------------------------
extern "C" void kernel_launch(void* const* d_in, const int* in_sizes, int n_in,
                              void* d_out, int out_size, void* d_ws, size_t ws_size,
                              hipStream_t stream)
{
    (void)in_sizes; (void)n_in; (void)out_size; (void)ws_size;
    const float* xr = (const float*)d_in[0];
    const float* xi = (const float*)d_in[1];
    const float* th = (const float*)d_in[2];
    const float* ph = (const float*)d_in[3];
    const float* ga = (const float*)d_in[4];

    char* ws = (char*)d_ws;
    _Float16* Bpack = (_Float16*)ws;                 // 3*2*65536 halves = 768 KB
    float* X1r = (float*)(ws + 786432);
    float* X1i = X1r + BATCH * N_WG;
    float* X2r = X1i + BATCH * N_WG;
    float* X2i = X2r + BATCH * N_WG;
    float* Ibuf = X1r;                               // reuse X1 for intensities

    compose_mesh<<<48, 256, 0, stream>>>(th, ph, ga, Bpack);
    mesh_gemm<<<512, 256, 0, stream>>>(xr,  xi,  Bpack + 0 * 131072, X1r, X1i, 0);
    mesh_gemm<<<512, 256, 0, stream>>>(X1r, X1i, Bpack + 1 * 131072, X2r, X2i, 0);
    mesh_gemm<<<512, 256, 0, stream>>>(X2r, X2i, Bpack + 2 * 131072, Ibuf, X2i, 1);
    normalize_k<<<1024, 256, 0, stream>>>(Ibuf, (float*)d_out);
}
